// DenseBipartiteGAT_69664369541130
// MI455X (gfx1250) — matrix operand order repaired and checked by reading, not compiled, not run on hardware
//
#include <hip/hip_runtime.h>

typedef __attribute__((ext_vector_type(16))) __bf16 v16bf;
typedef __attribute__((ext_vector_type(2)))  __bf16 v2bf;
typedef __attribute__((ext_vector_type(8)))  float  v8f;

#define NSRC 1024
#define NTGT 1024
#define BATCH 8
#define INCH 256
#define HEADS 4
#define HDIM 64
#define NROWS (BATCH * NSRC)

// ---------------- bf16 pair conversion ----------------
__device__ __forceinline__ unsigned int pack2(float a, float b) {
#if __has_builtin(__builtin_amdgcn_cvt_pk_bf16_f32)
    return __builtin_bit_cast(unsigned int, __builtin_amdgcn_cvt_pk_bf16_f32(a, b));
#else
    // native fptrunc: lowers to hardware v_cvt if the target has it
    v2bf v;
    v[0] = (__bf16)a;
    v[1] = (__bf16)b;
    return __builtin_bit_cast(unsigned int, v);
#endif
}

union Frag32 { int4 q[2]; unsigned int u[8]; v16bf v; };
union Q4 { unsigned int u[4]; int4 q; };

// ---------------- CDNA5 async global->LDS copy helpers ----------------------
__device__ __forceinline__ unsigned lds_off(const void* p) {
    return (unsigned)(size_t)p;   // generic LDS address: low 32 bits = LDS offset
}
__device__ __forceinline__ void async_b128(unsigned ldsoff, unsigned voff, const void* base) {
    asm volatile("global_load_async_to_lds_b128 %0, %1, %2"
                 :: "v"(ldsoff), "v"(voff), "s"(base) : "memory");
}
__device__ __forceinline__ void async_b32(unsigned ldsoff, unsigned voff, const void* base) {
    asm volatile("global_load_async_to_lds_b32 %0, %1, %2"
                 :: "v"(ldsoff), "v"(voff), "s"(base) : "memory");
}
__device__ __forceinline__ void wait_async0() {
#if __has_builtin(__builtin_amdgcn_s_wait_asynccnt)
    __builtin_amdgcn_s_wait_asynccnt(0);
#else
    asm volatile("s_wait_asynccnt 0" ::: "memory");
#endif
}

// ---------------------------------------------------------------------------
// Kernel P: elementwise f32 -> packed bf16 (8 floats / thread)
// ---------------------------------------------------------------------------
__global__ __launch_bounds__(256)
void gat_pack_kernel(const float* __restrict__ src,
                     unsigned int* __restrict__ dst, int n8) {
    int i = blockIdx.x * blockDim.x + threadIdx.x;
    if (i >= n8) return;
    const float* s = src + (size_t)i * 8;
    float4 f0 = *(const float4*)s;
    float4 f1 = *(const float4*)(s + 4);
    Q4 q;
    q.u[0] = pack2(f0.x, f0.y); q.u[1] = pack2(f0.z, f0.w);
    q.u[2] = pack2(f1.x, f1.y); q.u[3] = pack2(f1.z, f1.w);
    *(int4*)(dst + (size_t)i * 4) = q.q;
}

// ---------------------------------------------------------------------------
// Kernel A: fold attention vector into weights: u[h][c] = sum_d att[h,d]*W[h*64+d,c]
// ---------------------------------------------------------------------------
__global__ void gat_fold_kernel(const float* __restrict__ W,
                                const float* __restrict__ att,
                                float* __restrict__ u) {
    int idx = blockIdx.x * blockDim.x + threadIdx.x;
    if (idx >= HEADS * INCH) return;
    int h = idx >> 8, c = idx & 255;
    float s = 0.f;
    #pragma unroll 8
    for (int d = 0; d < HDIM; ++d)
        s += att[h * HDIM + d] * W[(size_t)(h * HDIM + d) * INCH + c];
    u[idx] = s;
}

// ---------------------------------------------------------------------------
// Kernel B: a[row][h] = mask[row] ? x[row].u[h] : -3e30  (one wave per row)
// ---------------------------------------------------------------------------
__global__ __launch_bounds__(256)
void gat_alpha_kernel(const float* __restrict__ X,
                      const float* __restrict__ u,
                      const unsigned char* __restrict__ mask,
                      float* __restrict__ a) {
    int lane = threadIdx.x & 31;
    int row  = (blockIdx.x * blockDim.x + threadIdx.x) >> 5;
    if (row >= NROWS) return;
    const float* x = X + (size_t)row * INCH + lane * 8;
    float4 x0 = *(const float4*)x;
    float4 x1 = *(const float4*)(x + 4);
    float acc[HEADS];
    #pragma unroll
    for (int h = 0; h < HEADS; ++h) {
        const float* uh = u + h * INCH + lane * 8;
        float4 u0 = *(const float4*)uh;
        float4 u1 = *(const float4*)(uh + 4);
        float s = x0.x*u0.x + x0.y*u0.y + x0.z*u0.z + x0.w*u0.w
                + x1.x*u1.x + x1.y*u1.y + x1.z*u1.z + x1.w*u1.w;
        #pragma unroll
        for (int o = 16; o > 0; o >>= 1) s += __shfl_xor(s, o, 32);
        acc[h] = s;
    }
    if (lane == 0) {
        float mk = mask[row] ? 0.f : -3e30f;
        Q4 q;
        q.u[0] = __float_as_uint(acc[0] + mk);
        q.u[1] = __float_as_uint(acc[1] + mk);
        q.u[2] = __float_as_uint(acc[2] + mk);
        q.u[3] = __float_as_uint(acc[3] + mk);
        *(int4*)&a[(size_t)row * HEADS] = q.q;
    }
}

// ---------------------------------------------------------------------------
// Kernel C: h_srcT[(b*4+h)*64+d][s] (bf16 pairs along s) = X @ W^T
// WMMA bf16; pre-packed bf16 tiles staged in LDS via double-buffered async
// copies; fragments loaded with ds_load_b128 (no conversion in the loop).
// Xu/Wu: packed bf16 (uint = 2 elems), row strides 128 uints.
// ---------------------------------------------------------------------------
__global__ __launch_bounds__(128)
void gat_proj_kernel(const unsigned int* __restrict__ Xu,
                     const unsigned int* __restrict__ Wu,
                     unsigned int* __restrict__ hTu) {
    __shared__ unsigned int AsU[2][64 * 16];   // [m][k/2] bf16 pairs, 4KB/buf
    __shared__ unsigned int BsU[2][64 * 16];   // [n][k/2]

    const int tid  = threadIdx.x;
    const int lane = tid & 31;
    const int wv   = tid >> 5;
    const int row0 = blockIdx.x * 64;   // 64 | 1024 -> tile within one batch
    const int h    = blockIdx.y;
    const int col0 = h * 64;

    // per-thread async staging: row r, 16-bf16 half (32 bytes = 2 x b128)
    const int r    = tid >> 1;
    const int half = (tid & 1) * 16;           // bf16 element offset
    const unsigned aV = (unsigned)(((row0 + r) * INCH + half) * 2);
    const unsigned bV = (unsigned)(((col0 + r) * INCH + half) * 2);
    unsigned aL[2], bL[2];
    aL[0] = lds_off(&AsU[0][r * 16 + (half >> 1)]);
    aL[1] = lds_off(&AsU[1][r * 16 + (half >> 1)]);
    bL[0] = lds_off(&BsU[0][r * 16 + (half >> 1)]);
    bL[1] = lds_off(&BsU[1][r * 16 + (half >> 1)]);

    auto issue = [&](int k0, int buf) {
        unsigned ko = (unsigned)k0 * 2;
        async_b128(aL[buf],      aV + ko,      Xu);
        async_b128(aL[buf] + 16, aV + ko + 16, Xu);
        async_b128(bL[buf],      bV + ko,      Wu);
        async_b128(bL[buf] + 16, bV + ko + 16, Wu);
    };

    v8f acc[4];
    #pragma unroll
    for (int i = 0; i < 4; ++i)
        #pragma unroll
        for (int j = 0; j < 8; ++j) acc[i][j] = 0.f;

    const int m   = lane & 15;
    const int kbA = (lane < 16) ? 0 : 4;    // uint offsets
    const int kbB = (lane < 16) ? 0 : 4;

    issue(0, 0);
    for (int c = 0; c < INCH / 32; ++c) {
        const int buf = c & 1;
        wait_async0();
        __syncthreads();
        if (c + 1 < INCH / 32) issue((c + 1) * 32, buf ^ 1);

        Frag32 af;
        const unsigned int* ap = &AsU[buf][(wv * 16 + m) * 16 + kbA];
        af.q[0] = *(const int4*)(ap);
        af.q[1] = *(const int4*)(ap + 8);
        #pragma unroll
        for (int nt = 0; nt < 4; ++nt) {
            Frag32 bf_;
            const unsigned int* bp = &BsU[buf][(nt * 16 + m) * 16 + kbB];
            bf_.q[0] = *(const int4*)(bp);
            bf_.q[1] = *(const int4*)(bp + 8);
            acc[nt] = __builtin_amdgcn_wmma_f32_16x16x32_bf16(
                false, af.v, false, bf_.v, (short)0, acc[nt], false, false);
        }
    }

    // transposed bf16 store: lane holds column d fixed, 8 consecutive s rows
    const int n      = lane & 15;
    const int srow0  = row0 + wv * 16 + ((lane < 16) ? 0 : 8);
    const int b      = row0 / NSRC;
    const int s_in_b = srow0 - b * NSRC;
    #pragma unroll
    for (int nt = 0; nt < 4; ++nt) {
        int din = nt * 16 + n;
        Q4 q;
        #pragma unroll
        for (int j = 0; j < 4; ++j) q.u[j] = pack2(acc[nt][2 * j], acc[nt][2 * j + 1]);
        size_t idx = ((size_t)(b * HEADS + h) * HDIM + din) * (NSRC / 2) + (s_in_b >> 1);
        *(int4*)(hTu + idx) = q.q;
    }
}

// ---------------------------------------------------------------------------
// Kernel D: fused masked softmax + aggregation, flash-style online softmax.
// Block = 4 waves = 4 heads x 16 targets; 32-source chunks streamed through
// LDS with double-buffered async copies.
// ---------------------------------------------------------------------------
__global__ __launch_bounds__(128)
void gat_attn_kernel(const float* __restrict__ adj,
                     const unsigned char* __restrict__ mask,
                     const float* __restrict__ a_src,
                     const float* __restrict__ a_tgt,
                     const unsigned int* __restrict__ hTu,
                     const float* __restrict__ bias,
                     float* __restrict__ out) {
    __shared__ float adjS[2][16 * 32];   // [t][k]
    __shared__ float aSm[2][HEADS * 32]; // [h][k], pre-masked a_src

    const int tid  = threadIdx.x;
    const int lane = tid & 31;
    const int h    = tid >> 5;
    const int b    = blockIdx.y;
    const int t0   = blockIdx.x * 16;

    // async staging addresses
    const int rr   = tid >> 3;
    const int part = (tid & 7) * 4;
    const unsigned adjV = (unsigned)((((b * NTGT + t0 + rr) * NSRC) + part) * 4);
    const int hh = tid >> 5, k2 = tid & 31;
    const unsigned aV = (unsigned)((((b * NSRC + k2) * HEADS) + hh) * 4);
    unsigned adjL[2], aL[2];
    adjL[0] = lds_off(&adjS[0][rr * 32 + part]); adjL[1] = lds_off(&adjS[1][rr * 32 + part]);
    aL[0]   = lds_off(&aSm[0][hh * 32 + k2]);    aL[1]   = lds_off(&aSm[1][hh * 32 + k2]);

    auto issue = [&](int i, int buf) {
        async_b128(adjL[buf], adjV + (unsigned)i * 128, adj);
        async_b32(aL[buf], aV + (unsigned)i * 512, a_src);
    };

    const int trow = lane & 15;
    const int kbA  = (lane < 16) ? 0 : 8;
    int kk[8];
    #pragma unroll
    for (int i = 0; i < 4; ++i) kk[i] = kbA + 2 * i;
    #pragma unroll
    for (int i = 4; i < 8; ++i) kk[i] = 16 + kbA + 2 * (i - 4);

    const float at = a_tgt[((size_t)b * NTGT + t0 + trow) * HEADS + h];  // pre-masked
    const float mt = mask[(size_t)b * NTGT + t0 + trow] ? 1.f : 0.f;

    v8f acc[4];
    #pragma unroll
    for (int i = 0; i < 4; ++i)
        #pragma unroll
        for (int j = 0; j < 8; ++j) acc[i][j] = 0.f;

    float m_run = -1e30f, l_run = 0.f;

    const int n   = lane & 15;
    const int kbB = (lane < 16) ? 0 : 8;
    const unsigned int* hbase = hTu + ((size_t)(b * HEADS + h) * HDIM) * (NSRC / 2);

    issue(0, 0);
    for (int it = 0; it < NSRC / 32; ++it) {
        const int buf = it & 1;
        const int s0  = it * 32;
        wait_async0();
        __syncthreads();
        if (it + 1 < NSRC / 32) issue(it + 1, buf ^ 1);

        // scores for this lane's 16 (t,s) cells (already WMMA A-layout ordered)
        float ev[16];
        float mx = -1e30f;
        #pragma unroll
        for (int i = 0; i < 8; ++i) {
            int k = kk[i];
            float2 av  = *(const float2*)&adjS[buf][trow * 32 + k];
            float2 asv = *(const float2*)&aSm[buf][h * 32 + k];
            float e0 = at + asv.x; e0 = (e0 >= 0.f) ? e0 : 0.2f * e0;
            float e1 = at + asv.y; e1 = (e1 >= 0.f) ? e1 : 0.2f * e1;
            ev[2 * i]     = (av.x != 0.f) ? e0 : -1e30f;
            ev[2 * i + 1] = (av.y != 0.f) ? e1 : -1e30f;
            mx = fmaxf(mx, fmaxf(ev[2 * i], ev[2 * i + 1]));
        }
        mx = fmaxf(mx, __shfl_xor(mx, 16, 32));
        float m_new = fmaxf(m_run, mx);
        float alpha = __expf(m_run - m_new);

        float lsum = 0.f;
        Frag32 af;
        #pragma unroll
        for (int i = 0; i < 8; ++i) {
            float p0 = (ev[2 * i]     < -1e29f) ? 0.f : __expf(ev[2 * i]     - m_new);
            float p1 = (ev[2 * i + 1] < -1e29f) ? 0.f : __expf(ev[2 * i + 1] - m_new);
            lsum += p0 + p1;
            af.u[i] = pack2(p0, p1);
        }
        lsum += __shfl_xor(lsum, 16, 32);
        l_run = l_run * alpha + lsum;
        m_run = m_new;

        // rescale accumulators by per-row alpha (lane permute broadcast)
        #pragma unroll
        for (int rx = 0; rx < 8; ++rx) {
            float am = __shfl(alpha, rx + ((lane < 16) ? 0 : 8), 32);
            #pragma unroll
            for (int nt = 0; nt < 4; ++nt) acc[nt][rx] *= am;
        }

        // B fragments straight from global hT (L2-hot), 2 x 16B per fragment
        #pragma unroll
        for (int nt = 0; nt < 4; ++nt) {
            const unsigned int* bp =
                hbase + (size_t)(nt * 16 + n) * (NSRC / 2) + ((s0 + kbB) >> 1);
            Frag32 bf_;
            bf_.q[0] = *(const int4*)(bp);
            bf_.q[1] = *(const int4*)(bp + 8);
            acc[nt] = __builtin_amdgcn_wmma_f32_16x16x32_bf16(
                false, af.v, false, bf_.v, (short)0, acc[nt], false, false);
        }
    }

    // epilogue: normalize rows, add bias, apply target mask, write out
    float invl = 1.f / (l_run + 1e-12f);
    float bval[4];
    #pragma unroll
    for (int nt = 0; nt < 4; ++nt) bval[nt] = bias[h * HDIM + nt * 16 + n];
    #pragma unroll
    for (int rx = 0; rx < 8; ++rx) {
        int mrow = rx + ((lane < 16) ? 0 : 8);
        float q = __shfl(invl, mrow, 32);
        float w = __shfl(mt,   mrow, 32);
        float* op = out + ((size_t)(b * NTGT + t0 + mrow)) * (HEADS * HDIM) + h * HDIM;
        #pragma unroll
        for (int nt = 0; nt < 4; ++nt)
            op[nt * 16 + n] = (acc[nt][rx] * q + bval[nt]) * w;
    }
}

// ---------------------------------------------------------------------------
extern "C" void kernel_launch(void* const* d_in, const int* in_sizes, int n_in,
                              void* d_out, int out_size, void* d_ws, size_t ws_size,
                              hipStream_t stream) {
    const float* x_src   = (const float*)d_in[0];
    const float* x_tgt   = (const float*)d_in[1];
    const float* adj     = (const float*)d_in[2];
    const unsigned char* mask = (const unsigned char*)d_in[3];
    const float* W_src   = (const float*)d_in[4];
    const float* W_tgt   = (const float*)d_in[5];
    const float* att_src = (const float*)d_in[6];
    const float* att_tgt = (const float*)d_in[7];
    const float* bias    = (const float*)d_in[8];
    float* out = (float*)d_out;

    char* ws = (char*)d_ws;
    float* u_src = (float*)(ws);                              //   4 KB
    float* u_tgt = (float*)(ws + (4 << 10));                  //   4 KB
    float* a_src = (float*)(ws + (8 << 10));                  // 128 KB (pre-masked)
    float* a_tgt = (float*)(ws + (8 << 10) + (128 << 10));    // 128 KB
    unsigned int* hTu = (unsigned int*)(ws + (8 << 10) + (256 << 10));          // 4 MB
    unsigned int* Xu  = (unsigned int*)(ws + (8 << 10) + (256 << 10) + (4 << 20)); // 4 MB
    unsigned int* Wu  = (unsigned int*)(ws + (8 << 10) + (256 << 10) + (8 << 20)); // 128 KB

    gat_fold_kernel<<<4, 256, 0, stream>>>(W_src, att_src, u_src);
    gat_fold_kernel<<<4, 256, 0, stream>>>(W_tgt, att_tgt, u_tgt);

    gat_alpha_kernel<<<NROWS / 8, 256, 0, stream>>>(x_src, u_src, mask, a_src);
    gat_alpha_kernel<<<NROWS / 8, 256, 0, stream>>>(x_tgt, u_tgt, mask, a_tgt);

    gat_pack_kernel<<<(NROWS * INCH / 8) / 256, 256, 0, stream>>>(x_src, Xu, NROWS * INCH / 8);
    gat_pack_kernel<<<(INCH * INCH / 8 + 255) / 256, 256, 0, stream>>>(W_src, Wu, INCH * INCH / 8);

    dim3 gproj(NROWS / 64, HEADS);
    gat_proj_kernel<<<gproj, 128, 0, stream>>>(Xu, Wu, hTu);

    dim3 gattn(NTGT / 16, BATCH);
    gat_attn_kernel<<<gattn, 128, 0, stream>>>(adj, mask, a_src, a_tgt, hTu, bias, out);
}
